// NTXent_45423574123183
// MI455X (gfx1250) — matrix operand order repaired and checked by reading, not compiled, not run on hardware
//
#include <hip/hip_runtime.h>
#include <math.h>

#define B_SZ   4096
#define D_DIM  128
#define N2     8192          // 2*B
#define INV_TAU 2.0f         // 1/0.5

typedef _Float16 v16h __attribute__((ext_vector_type(16)));
typedef _Float16 v8h  __attribute__((ext_vector_type(8)));
typedef _Float16 v4h  __attribute__((ext_vector_type(4)));
typedef float    v8f  __attribute__((ext_vector_type(8)));

union Frag16 { v16h v; v8h h[2]; };

// ---------------------------------------------------------------------------
// Zero the scalar output (we accumulate into it with atomics).
// ---------------------------------------------------------------------------
__global__ void ntx_init_kernel(float* out) {
    if (threadIdx.x == 0 && blockIdx.x == 0) out[0] = 0.0f;
}

// ---------------------------------------------------------------------------
// L2-normalize rows of e_i / e_j into a packed f16 reps buffer [2B, 128].
// One wave32 per row: each lane handles 4 contiguous floats.
// ---------------------------------------------------------------------------
__global__ void ntx_norm_kernel(const float* __restrict__ e_i,
                                const float* __restrict__ e_j,
                                _Float16* __restrict__ reps) {
    const int lane = threadIdx.x & 31;
    const int row  = blockIdx.x * 8 + (threadIdx.x >> 5);   // [0, 2B)
    const float* src = (row < B_SZ) ? (e_i + (size_t)row * D_DIM)
                                    : (e_j + (size_t)(row - B_SZ) * D_DIM);
    float4 x = reinterpret_cast<const float4*>(src)[lane];
    float ss = x.x * x.x + x.y * x.y + x.z * x.z + x.w * x.w;
    #pragma unroll
    for (int off = 16; off >= 1; off >>= 1)
        ss += __shfl_xor(ss, off, 32);
    const float scale = 1.0f / fmaxf(sqrtf(ss), 1e-12f);
    v4h h;
    h[0] = (_Float16)(x.x * scale);
    h[1] = (_Float16)(x.y * scale);
    h[2] = (_Float16)(x.z * scale);
    h[3] = (_Float16)(x.w * scale);
    *reinterpret_cast<v4h*>(reps + (size_t)row * D_DIM + lane * 4) = h;
}

// ---------------------------------------------------------------------------
// Positive pairs: pos_r = z_i[r] . z_j[r] (appears twice in the loss).
// One wave32 per row r in [0, B). Contribution: -2*pos_r/(tau*2B).
// ---------------------------------------------------------------------------
__global__ void ntx_pos_kernel(const _Float16* __restrict__ reps,
                               float* __restrict__ out) {
    const int lane = threadIdx.x & 31;
    const int row  = blockIdx.x * 8 + (threadIdx.x >> 5);   // [0, B)
    const _Float16* zi = reps + (size_t)row * D_DIM;
    const _Float16* zj = reps + (size_t)(row + B_SZ) * D_DIM;
    float s = 0.0f;
    #pragma unroll
    for (int t = 0; t < 4; ++t) {
        int idx = lane * 4 + t;
        s += (float)zi[idx] * (float)zj[idx];
    }
    #pragma unroll
    for (int off = 16; off >= 1; off >>= 1)
        s += __shfl_xor(s, off, 32);
    if (lane == 0)
        atomicAdd(out, -s * (INV_TAU / (float)B_SZ));   // -2*pos/(tau*2B)
}

// ---------------------------------------------------------------------------
// Fused sim-GEMM + exp + row-sum + log. Block = 8 waves owns 16 rows and
// sweeps all 8192 columns; sim tile 16x16 computed with 4 chained
// v_wmma_f32_16x16x32_f16 over K = D = 128. Never materializes sim.
// ---------------------------------------------------------------------------
__global__ void __launch_bounds__(256)
ntx_simloss_kernel(const _Float16* __restrict__ reps,
                   float* __restrict__ out) {
    const int lane    = threadIdx.x & 31;
    const int wave    = threadIdx.x >> 5;       // 0..7
    const int m       = lane & 15;              // row/col slot within tile
    const int hi      = lane >> 4;              // lane-group (0 or 1)
    const int rowbase = blockIdx.x * 16;        // 512 blocks cover 8192 rows

    // --- Preload A fragments (16 rows x K=128), reused for all 64 col tiles.
    // 16-bit A 16x32 layout: lanes0-15 -> K[0..7]+K[16..23]; lanes16-31 -> +8.
    const _Float16* arow = reps + (size_t)(rowbase + m) * D_DIM;
    Frag16 a[4];
    #pragma unroll
    for (int k = 0; k < 4; ++k) {
        const int base = k * 32 + hi * 8;
        a[k].h[0] = *reinterpret_cast<const v8h*>(arow + base);
        a[k].h[1] = *reinterpret_cast<const v8h*>(arow + base + 16);
    }

    float acc[8];
    #pragma unroll
    for (int j = 0; j < 8; ++j) acc[j] = 0.0f;

    // --- Sweep all columns: 8 waves x 16 cols = 128 cols/iter, 64 iters.
    for (int it = 0; it < 64; ++it) {
        const int colbase = (it * 8 + wave) * 16;
        // B 32x16 layout: lane holds column (m) = reps row colbase+m,
        // K[0..7]+K[16..23] for lanes0-15; lanes16-31 -> +8 (matches A).
        const _Float16* brow = reps + (size_t)(colbase + m) * D_DIM;
        v8f c = {};
        #pragma unroll
        for (int k = 0; k < 4; ++k) {
            Frag16 b;
            const int kb = k * 32 + hi * 8;
            b.h[0] = *reinterpret_cast<const v8h*>(brow + kb);
            b.h[1] = *reinterpret_cast<const v8h*>(brow + kb + 16);
            c = __builtin_amdgcn_wmma_f32_16x16x32_f16(
                    /*neg_a=*/false, a[k].v, /*neg_b=*/false, b.v,
                    /*c_mod=*/(short)0, c, /*reuse_a=*/false, /*reuse_b=*/false);
        }
        // C/D layout: lane = col m, VGPR j = row j + 8*hi.
        #pragma unroll
        for (int j = 0; j < 8; ++j)
            acc[j] += __expf(c[j] * INV_TAU);
    }

    // --- Reduce 8 per-lane row partials across the 16 column-lanes (LDS),
    //     and across the block's 8 waves, into full row denominators.
    __shared__ float rowsum[16];
    if (threadIdx.x < 16) rowsum[threadIdx.x] = 0.0f;
    __syncthreads();
    #pragma unroll
    for (int j = 0; j < 8; ++j)
        atomicAdd(&rowsum[j + 8 * hi], acc[j]);
    __syncthreads();

    if (threadIdx.x < 16) {
        // remove self-similarity term exp(sim[r,r]/tau) = e^{1/tau}
        float denom = rowsum[threadIdx.x] - __expf(INV_TAU);
        atomicAdd(out, __logf(denom) * (1.0f / (float)N2));
    }
}

// ---------------------------------------------------------------------------
extern "C" void kernel_launch(void* const* d_in, const int* in_sizes, int n_in,
                              void* d_out, int out_size, void* d_ws, size_t ws_size,
                              hipStream_t stream) {
    (void)in_sizes; (void)n_in; (void)out_size; (void)ws_size;
    const float* e_i = (const float*)d_in[0];
    const float* e_j = (const float*)d_in[1];
    float*       out  = (float*)d_out;
    _Float16*    reps = (_Float16*)d_ws;   // [2B, 128] f16 = 2 MB

    ntx_init_kernel<<<1, 64, 0, stream>>>(out);
    ntx_norm_kernel<<<N2 / 8, 256, 0, stream>>>(e_i, e_j, reps);    // 8 rows/block
    ntx_pos_kernel<<<B_SZ / 8, 256, 0, stream>>>(reps, out);
    ntx_simloss_kernel<<<N2 / 16, 256, 0, stream>>>(reps, out);     // 512 blocks
}
